// LSTM_44564580663750
// MI455X (gfx1250) — matrix operand order repaired and checked by reading, not compiled, not run on hardware
//
#include <hip/hip_runtime.h>
#include <hip/hip_bf16.h>

typedef __attribute__((ext_vector_type(16))) _Float16 v16h;
typedef __attribute__((ext_vector_type(8)))  _Float16 v8h;
typedef __attribute__((ext_vector_type(4)))  _Float16 v4h;
typedef __attribute__((ext_vector_type(8)))  float    v8f;

#define BB 256
#define TT 512
#define II 200
#define H1 64
#define H2 32

// padded LDS row strides
#define HS1 72    // h rows, halves (144B, 16B-aligned)
#define HS2 40    // h rows, halves (80B, 16B-aligned)
#define LDG1 260  // gate rows, floats
#define LDG2 132  // gate rows, floats

__device__ __forceinline__ float sigmoidf_(float x) {
    return 1.0f / (1.0f + __expf(-x));
}
__device__ __forceinline__ float tanhf_(float x) {
    return 1.0f - 2.0f / (__expf(2.0f * x) + 1.0f);
}

// Load 8 contiguous f32 (16B-aligned) and convert into v[base..base+7].
__device__ __forceinline__ void ld8(v16h& v, const int base, const float* __restrict__ p) {
    const float4 f0 = *reinterpret_cast<const float4*>(p);
    const float4 f1 = *reinterpret_cast<const float4*>(p + 4);
    v[base + 0] = (_Float16)f0.x; v[base + 1] = (_Float16)f0.y;
    v[base + 2] = (_Float16)f0.z; v[base + 3] = (_Float16)f0.w;
    v[base + 4] = (_Float16)f1.x; v[base + 5] = (_Float16)f1.y;
    v[base + 6] = (_Float16)f1.z; v[base + 7] = (_Float16)f1.w;
}

// f32 -> f16 weight pre-conversion
__global__ __launch_bounds__(256)
void f32_to_f16_kernel(const float* __restrict__ src, _Float16* __restrict__ dst, int n) {
    const int i = blockIdx.x * 256 + threadIdx.x;
    if (i < n) dst[i] = (_Float16)src[i];
}

// ---------------------------------------------------------------------------
// Input-projection GEMM: Y[m, n] = sum_k X[row(m), k]*W16[n, k] + b1[n] + b2[n]
// One wave per block -> 16(M) x 128(N) tile; f16 WMMA; f32 accumulate.
// W16 pre-converted to f16 -> B fragments are raw b128 loads, no cvt.
// REMAP: m = t*BB + b maps to input row b*TT + t (layer-1 input layout).
// ---------------------------------------------------------------------------
template <int K, bool REMAP>
__global__ __launch_bounds__(32)
void proj_gemm_kernel(const float* __restrict__ X,
                      const _Float16* __restrict__ W16,  // (N_total, K) row-major f16
                      const float* __restrict__ b1,
                      const float* __restrict__ b2,
                      float* __restrict__ Y, int ldy)
{
    const int lane = threadIdx.x & 31;
    const int half = lane >> 4;
    const int mr   = lane & 15;
    const int m0   = blockIdx.x * 16;
    const int n0   = blockIdx.y * 128;
    const int m    = m0 + mr;

    size_t xoff;
    if constexpr (REMAP) {
        const int b = m & (BB - 1);
        const int t = m >> 8;
        xoff = ((size_t)b * TT + t) * K;
    } else {
        xoff = (size_t)m * K;
    }
    const float* __restrict__ xrow = X + xoff;

    v8f acc[8] = {};
    constexpr int KF = (K / 32) * 32;

    for (int k0 = 0; k0 < KF; k0 += 32) {
        v16h a;
        ld8(a, 0, xrow + k0 + half * 16);
        ld8(a, 8, xrow + k0 + half * 16 + 8);
#pragma unroll
        for (int nt = 0; nt < 8; ++nt) {
            const _Float16* wrow = W16 + (size_t)(n0 + nt * 16 + mr) * K + k0 + half * 16;
            const v16h b = *reinterpret_cast<const v16h*>(wrow);
            acc[nt] = __builtin_amdgcn_wmma_f32_16x16x32_f16(
                false, a, false, b, (short)0, acc[nt], false, false);
        }
    }

    if constexpr (K != KF) {
        static_assert(K - KF == 8, "tail handling assumes K % 32 == 8");
        const bool lo = (half == 0);
        const _Float16 z = (_Float16)0.0f;
        // A tail: 8 valid k on lane-half 0, rest zero (branch-free)
        v16h a;
        {
            const float4 f0 = *reinterpret_cast<const float4*>(xrow + KF);
            const float4 f1 = *reinterpret_cast<const float4*>(xrow + KF + 4);
            a[0] = lo ? (_Float16)f0.x : z; a[1] = lo ? (_Float16)f0.y : z;
            a[2] = lo ? (_Float16)f0.z : z; a[3] = lo ? (_Float16)f0.w : z;
            a[4] = lo ? (_Float16)f1.x : z; a[5] = lo ? (_Float16)f1.y : z;
            a[6] = lo ? (_Float16)f1.z : z; a[7] = lo ? (_Float16)f1.w : z;
#pragma unroll
            for (int j = 8; j < 16; ++j) a[j] = z;
        }
#pragma unroll
        for (int nt = 0; nt < 8; ++nt) {
            const v8h wt = *reinterpret_cast<const v8h*>(
                W16 + (size_t)(n0 + nt * 16 + mr) * K + KF);
            v16h b;
#pragma unroll
            for (int j = 0; j < 8; ++j) b[j] = lo ? wt[j] : z;
#pragma unroll
            for (int j = 8; j < 16; ++j) b[j] = z;
            acc[nt] = __builtin_amdgcn_wmma_f32_16x16x32_f16(
                false, a, false, b, (short)0, acc[nt], false, false);
        }
    }

    // store C/D: VGPR r -> row m0 + r + 8*half, col n0 + nt*16 + (lane&15)
#pragma unroll
    for (int nt = 0; nt < 8; ++nt) {
        const int n = n0 + nt * 16 + mr;
        const float bias = b1[n] + b2[n];
#pragma unroll
        for (int r = 0; r < 8; ++r) {
            const int mm = m0 + r + half * 8;
            Y[(size_t)mm * ldy + n] = acc[nt][r] + bias;
        }
    }
}

// ---------------------------------------------------------------------------
// Layer-1 recurrence: 16 blocks (16 batch rows) x 8 waves; 2 gate tiles/wave.
// h in LDS as f16 (A-frags = raw ds_load_b128), c in registers,
// Whh fragments in VGPRs, t-major xp (contiguous 16KB/step, prefetched).
// ---------------------------------------------------------------------------
__global__ __launch_bounds__(256)
void lstm1_kernel(const float* __restrict__ xp,    // (T*B, 256)
                  const float* __restrict__ Whh,   // (256, 64)
                  float* __restrict__ out1)        // (T*B, 64)
{
    __shared__ _Float16 h_lds[16 * HS1];
    __shared__ float    g_lds[16 * LDG1];

    const int tid  = threadIdx.x;
    const int lane = tid & 31;
    const int wave = tid >> 5;        // 0..7
    const int half = lane >> 4;
    const int mr   = lane & 15;
    const int b0   = blockIdx.x * 16;

    for (int i = tid; i < 16 * HS1; i += 256) h_lds[i] = (_Float16)0.0f;

    // preload recurrent-weight B fragments: 2 tiles x 2 K-slices
    v16h bf[2][2];
#pragma unroll
    for (int t2 = 0; t2 < 2; ++t2) {
        const float* wrow = Whh + (size_t)((wave * 2 + t2) * 16 + mr) * H1;
#pragma unroll
        for (int ks = 0; ks < 2; ++ks) {
            ld8(bf[t2][ks], 0, wrow + ks * 32 + half * 8);
            ld8(bf[t2][ks], 8, wrow + ks * 32 + 16 + half * 8);
        }
    }

    // per-thread cell state: element (mu, j0..j0+3)
    const int mu = tid >> 4;          // 0..15
    const int j0 = (tid & 15) * 4;    // 0..60
    float cc[4] = {0.0f, 0.0f, 0.0f, 0.0f};

    __syncthreads();

    for (int t = 0; t < TT; ++t) {
        const float* __restrict__ xrow = xp + ((size_t)t * BB + b0) * 256;
        if (t + 1 < TT)
            __builtin_prefetch(xp + ((size_t)(t + 1) * BB + b0) * 256 + tid * 16, 0, 3);

        // A fragments of h (16x64, two K-slices) straight from f16 LDS
        const v8h* hp = reinterpret_cast<const v8h*>(h_lds + mr * HS1);
        const v8h c0 = hp[half];
        const v8h c1 = hp[2 + half];
        const v8h c2 = hp[4 + half];
        const v8h c3 = hp[6 + half];
        const v16h a0 = __builtin_shufflevector(c0, c1, 0,1,2,3,4,5,6,7,8,9,10,11,12,13,14,15);
        const v16h a1 = __builtin_shufflevector(c2, c3, 0,1,2,3,4,5,6,7,8,9,10,11,12,13,14,15);

#pragma unroll
        for (int t2 = 0; t2 < 2; ++t2) {
            v8f acc = {};
            acc = __builtin_amdgcn_wmma_f32_16x16x32_f16(
                false, a0, false, bf[t2][0], (short)0, acc, false, false);
            acc = __builtin_amdgcn_wmma_f32_16x16x32_f16(
                false, a1, false, bf[t2][1], (short)0, acc, false, false);
            const int n = (wave * 2 + t2) * 16 + mr;
#pragma unroll
            for (int r = 0; r < 8; ++r)
                g_lds[(r + half * 8) * LDG1 + n] = acc[r];
        }
        __syncthreads();

        // fused gates + cell update; thread -> (mu, j0..j0+3), vectorized
        {
            const float* gr = g_lds + mu * LDG1 + j0;
            const float* xr = xrow + mu * 256 + j0;
            const float4 gi = *(const float4*)(gr);       const float4 xi = *(const float4*)(xr);
            const float4 gf = *(const float4*)(gr + 64);  const float4 xf = *(const float4*)(xr + 64);
            const float4 gg = *(const float4*)(gr + 128); const float4 xg = *(const float4*)(xr + 128);
            const float4 go = *(const float4*)(gr + 192); const float4 xo = *(const float4*)(xr + 192);
            const float iv[4] = {gi.x + xi.x, gi.y + xi.y, gi.z + xi.z, gi.w + xi.w};
            const float fv[4] = {gf.x + xf.x, gf.y + xf.y, gf.z + xf.z, gf.w + xf.w};
            const float gv[4] = {gg.x + xg.x, gg.y + xg.y, gg.z + xg.z, gg.w + xg.w};
            const float ov[4] = {go.x + xo.x, go.y + xo.y, go.z + xo.z, go.w + xo.w};
            float4 hv;
            float hvv[4];
            v4h hh;
#pragma unroll
            for (int q = 0; q < 4; ++q) {
                const float cv = sigmoidf_(fv[q]) * cc[q] + sigmoidf_(iv[q]) * tanhf_(gv[q]);
                cc[q] = cv;
                hvv[q] = sigmoidf_(ov[q]) * tanhf_(cv);
                hh[q] = (_Float16)hvv[q];
            }
            hv.x = hvv[0]; hv.y = hvv[1]; hv.z = hvv[2]; hv.w = hvv[3];
            *(v4h*)(h_lds + mu * HS1 + j0) = hh;
            *(float4*)(out1 + ((size_t)t * BB + b0 + mu) * H1 + j0) = hv;
        }
        __syncthreads();
    }
}

// ---------------------------------------------------------------------------
// Layer-2 recurrence + FC + softmax: 16 blocks x 4 waves; 128 gates.
// ---------------------------------------------------------------------------
__global__ __launch_bounds__(128)
void lstm2_kernel(const float* __restrict__ xp,    // (T*B, 128)
                  const float* __restrict__ Whh,   // (128, 32)
                  const float* __restrict__ fc_w,  // (2, 32)
                  const float* __restrict__ fc_b,  // (2,)
                  float* __restrict__ logits)      // (B, 2)
{
    __shared__ _Float16 h_lds[16 * HS2];
    __shared__ float    g_lds[16 * LDG2];

    const int tid  = threadIdx.x;
    const int lane = tid & 31;
    const int wave = tid >> 5;        // 0..3
    const int half = lane >> 4;
    const int mr   = lane & 15;
    const int b0   = blockIdx.x * 16;

    for (int i = tid; i < 16 * HS2; i += 128) h_lds[i] = (_Float16)0.0f;

    v16h bf[2];
#pragma unroll
    for (int t2 = 0; t2 < 2; ++t2) {
        const float* wrow = Whh + (size_t)((wave * 2 + t2) * 16 + mr) * H2;
        ld8(bf[t2], 0, wrow + half * 8);
        ld8(bf[t2], 8, wrow + 16 + half * 8);
    }

    const int mu = tid >> 3;          // 0..15
    const int j0 = (tid & 7) * 4;     // 0..28
    float cc[4] = {0.0f, 0.0f, 0.0f, 0.0f};

    __syncthreads();

    for (int t = 0; t < TT; ++t) {
        const float* __restrict__ xrow = xp + ((size_t)t * BB + b0) * 128;
        if (t + 1 < TT)
            __builtin_prefetch(xp + ((size_t)(t + 1) * BB + b0) * 128 + tid * 16, 0, 3);

        const v8h* hp = reinterpret_cast<const v8h*>(h_lds + mr * HS2);
        const v8h c0 = hp[half];
        const v8h c1 = hp[2 + half];
        const v16h a = __builtin_shufflevector(c0, c1, 0,1,2,3,4,5,6,7,8,9,10,11,12,13,14,15);

#pragma unroll
        for (int t2 = 0; t2 < 2; ++t2) {
            v8f acc = {};
            acc = __builtin_amdgcn_wmma_f32_16x16x32_f16(
                false, a, false, bf[t2], (short)0, acc, false, false);
            const int n = (wave * 2 + t2) * 16 + mr;
#pragma unroll
            for (int r = 0; r < 8; ++r)
                g_lds[(r + half * 8) * LDG2 + n] = acc[r];
        }
        __syncthreads();

        {
            const float* gr = g_lds + mu * LDG2 + j0;
            const float* xr = xrow + mu * 128 + j0;
            const float4 gi = *(const float4*)(gr);      const float4 xi = *(const float4*)(xr);
            const float4 gf = *(const float4*)(gr + 32); const float4 xf = *(const float4*)(xr + 32);
            const float4 gg = *(const float4*)(gr + 64); const float4 xg = *(const float4*)(xr + 64);
            const float4 go = *(const float4*)(gr + 96); const float4 xo = *(const float4*)(xr + 96);
            const float iv[4] = {gi.x + xi.x, gi.y + xi.y, gi.z + xi.z, gi.w + xi.w};
            const float fv[4] = {gf.x + xf.x, gf.y + xf.y, gf.z + xf.z, gf.w + xf.w};
            const float gv[4] = {gg.x + xg.x, gg.y + xg.y, gg.z + xg.z, gg.w + xg.w};
            const float ov[4] = {go.x + xo.x, go.y + xo.y, go.z + xo.z, go.w + xo.w};
            v4h hh;
#pragma unroll
            for (int q = 0; q < 4; ++q) {
                const float cv = sigmoidf_(fv[q]) * cc[q] + sigmoidf_(iv[q]) * tanhf_(gv[q]);
                cc[q] = cv;
                hh[q] = (_Float16)(sigmoidf_(ov[q]) * tanhf_(cv));
            }
            *(v4h*)(h_lds + mu * HS2 + j0) = hh;
        }
        __syncthreads();
    }

    // FC (2 classes) + softmax on the final hidden state
    if (tid < 16) {
        const _Float16* hv = h_lds + tid * HS2;
        float l0 = fc_b[0], l1 = fc_b[1];
#pragma unroll
        for (int j = 0; j < H2; ++j) {
            const float h = (float)hv[j];
            l0 += h * fc_w[j];
            l1 += h * fc_w[H2 + j];
        }
        const float mx = fmaxf(l0, l1);
        const float e0 = __expf(l0 - mx), e1 = __expf(l1 - mx);
        const float inv = 1.0f / (e0 + e1);
        logits[(b0 + tid) * 2 + 0] = e0 * inv;
        logits[(b0 + tid) * 2 + 1] = e1 * inv;
    }
}

extern "C" void kernel_launch(void* const* d_in, const int* in_sizes, int n_in,
                              void* d_out, int out_size, void* d_ws, size_t ws_size,
                              hipStream_t stream) {
    (void)in_sizes; (void)n_in; (void)out_size; (void)ws_size;

    const float* x    = (const float*)d_in[0];
    const float* Wih1 = (const float*)d_in[1];
    const float* Whh1 = (const float*)d_in[2];
    const float* bih1 = (const float*)d_in[3];
    const float* bhh1 = (const float*)d_in[4];
    const float* Wih2 = (const float*)d_in[5];
    const float* Whh2 = (const float*)d_in[6];
    const float* bih2 = (const float*)d_in[7];
    const float* bhh2 = (const float*)d_in[8];
    const float* fc_w = (const float*)d_in[9];
    const float* fc_b = (const float*)d_in[10];
    float* out = (float*)d_out;

    const size_t rows = (size_t)BB * TT;             // 131072
    float* xp1  = (float*)d_ws;                      // rows * 256 f32
    float* out1 = xp1 + rows * 256;                  // rows * 64  f32
    float* xp2  = xp1;                               // reuse dead xp1 region
    _Float16* w16a = (_Float16*)(out1 + rows * 64);  // 4*H1*II halves
    _Float16* w16b = w16a + 4 * H1 * II;             // 4*H2*H1 halves

    // Pre-convert projection weights to f16
    f32_to_f16_kernel<<<(4 * H1 * II + 255) / 256, 256, 0, stream>>>(Wih1, w16a, 4 * H1 * II);
    f32_to_f16_kernel<<<(4 * H2 * H1 + 255) / 256, 256, 0, stream>>>(Wih2, w16b, 4 * H2 * H1);

    // Layer-1 input projection (remaps (b,t)-major input to t-major rows)
    proj_gemm_kernel<II, true><<<dim3(rows / 16, 2), 32, 0, stream>>>(
        x, w16a, bih1, bhh1, xp1, 4 * H1);

    // Layer-1 recurrence
    lstm1_kernel<<<16, 256, 0, stream>>>(xp1, Whh1, out1);

    // Layer-2 input projection
    proj_gemm_kernel<H1, false><<<dim3(rows / 16, 1), 32, 0, stream>>>(
        out1, w16b, bih2, bhh2, xp2, 4 * H2);

    // Layer-2 recurrence + FC + softmax
    lstm2_kernel<<<16, 128, 0, stream>>>(xp2, Whh2, fc_w, fc_b, out);
}
